// FusionAttention_49538152792907
// MI455X (gfx1250) — matrix operand order repaired and checked by reading, not compiled, not run on hardware
//
#include <hip/hip_runtime.h>

// ---------------------------------------------------------------------------
// MI455X (gfx1250) attention+MLP pipeline.
//  * every GEMM -> V_WMMA_F32_16X16X32_BF16 (wave32, f32 accum)
//  * block tiles staged into LDS with GLOBAL_LOAD_ASYNC_TO_LDS_B128,
//    double-buffered and synchronized with s_wait_asynccnt (ASYNCcnt)
// ---------------------------------------------------------------------------

typedef __bf16 bf16_t;
typedef bf16_t v16bf __attribute__((ext_vector_type(16)));
typedef bf16_t v8bf  __attribute__((ext_vector_type(8)));
typedef float  v8f   __attribute__((ext_vector_type(8)));

#define B_   16
#define S_   2048
#define DIN  512
#define DFF  2048
#define LN_EPS 1e-5f
// SCALER = D_OUT/2 = 256  ->  alpha = 1/256 folded into the q*k^T GEMM

#define TPAD  80                  // LDS bytes per 32-elem bf16 tile row (64B data + 16B pad)
#define TILEB (128 * TPAD)        // one 128x32 bf16 tile in LDS (10 KB)

// ---------------- elementwise f32 -> bf16 ----------------
__global__ void __launch_bounds__(256)
cvt_f32_bf16(const float* __restrict__ in, bf16_t* __restrict__ out, int n) {
  int i = blockIdx.x * 256 + threadIdx.x;
  if (i < n) out[i] = (bf16_t)in[i];
}

// ---------------- transpose-convert W f32[K][N] -> bf16[N][K] ----------------
__global__ void __launch_bounds__(256)
wtrans(const float* __restrict__ in, bf16_t* __restrict__ out, int K, int N) {
  int i = blockIdx.x * 256 + threadIdx.x;
  if (i >= K * N) return;
  int n = i / K, k = i % K;               // output-coalesced
  out[i] = (bf16_t)in[k * N + n];
}

// ---------------- transpose v bf16 [b][t][d] -> [b][d][t] ----------------
__global__ void __launch_bounds__(256)
vtrans(const bf16_t* __restrict__ v, bf16_t* __restrict__ vt) {
  long i = (long)blockIdx.x * 256 + threadIdx.x;      // over B*DIN*S
  long per = (long)DIN * S_;
  long b = i / per; long r = i % per;
  long d = r / S_;  long t = r % S_;
  vt[i] = v[b * per + t * DIN + d];
}

// ---------------- async cooperative tile stage: global -> LDS ---------------
// 128 rows x 32 bf16 (64B data) per tile = 512 x 16B chunks;
// 8 waves x 2 instructions x 32 lanes, one b128 async copy per lane.
__device__ __forceinline__ void async_tile(const bf16_t* __restrict__ gbase,
                                           int ld, unsigned ldsbase,
                                           int w, int lane) {
#pragma unroll
  for (int t = 0; t < 2; ++t) {
    int c   = w * 64 + t * 32 + lane;       // chunk id 0..511
    int row = c >> 2, sub = c & 3;
    const bf16_t* gp = gbase + (long)row * ld + sub * 8;
    unsigned lp = ldsbase + (unsigned)(row * TPAD + sub * 16);
    asm volatile("global_load_async_to_lds_b128 %0, %1, off"
                 :: "v"(lp), "v"(gp) : "memory");
  }
}

// ---------------- WMMA fragment reads from LDS (ISA 7.12.2 layouts) ---------
// A (16x32): lane(0..15)=M, half=lane>>4; VGPR0..3: K=half*8+0..7; VGPR4..7: +16
__device__ __forceinline__ v16bf ldsA(const char* __restrict__ smA, int row, int lane) {
  int m = lane & 15, half = lane >> 4;
  const char* p = smA + (row + m) * TPAD + half * 16;
  union { v16bf v; v8bf h[2]; } u;
  u.h[0] = *(const v8bf*)(p);
  u.h[1] = *(const v8bf*)(p + 32);
  return u.v;
}
// B (32x16) from [N][K] tile: lane(0..15)=N, half=lane>>4; VGPR0..3: K=half*8+0..7; VGPR4..7: +16
__device__ __forceinline__ v16bf ldsB(const char* __restrict__ smB, int col, int lane) {
  int n = lane & 15, half = lane >> 4;
  const char* p = smB + (col + n) * TPAD + half * 16;
  union { v16bf v; v8bf h[2]; } u;
  u.h[0] = *(const v8bf*)(p);
  u.h[1] = *(const v8bf*)(p + 32);
  return u.v;
}

// ---------------- generic bf16 GEMM:  C[M][N] = alpha * A[M][K] * Bt[N][K]^T
// 256 threads = 8 waves; wave tile 32x64 (2x4 frags); block tile 128x128.
// Double-buffered LDS staging via async loads; WMMA from LDS.
__global__ void __launch_bounds__(256)
gemm_bf16(const bf16_t* __restrict__ A, const bf16_t* __restrict__ Bt,
          float* __restrict__ Cf, bf16_t* __restrict__ Cb,
          int K, int lda, int ldb, int ldc,
          long sA, long sB, long sC, float alpha) {
  __shared__ char smem[4 * TILEB];          // [buf0: A,B][buf1: A,B] = 40 KB
  const int tid  = threadIdx.x;
  const int lane = tid & 31;
  const int w    = tid >> 5;
  const int wm   = w >> 1;                  // 0..3
  const int wn   = w & 1;                   // 0..1
  const long row0 = (long)blockIdx.y * 128;
  const long col0 = (long)blockIdx.x * 128;
  const bf16_t* Abase = A  + (long)blockIdx.z * sA + row0 * lda;
  const bf16_t* Bbase = Bt + (long)blockIdx.z * sB + col0 * ldb;
  const unsigned smbase = (unsigned)(size_t)(const void*)smem; // LDS offset = low 32 bits of flat shared addr

  v8f acc[2][4] = {};
  const int nk = K / 32;

  // prologue: stage k-block 0 into buffer 0
  async_tile(Abase, lda, smbase, w, lane);
  async_tile(Bbase, ldb, smbase + TILEB, w, lane);

  int cur = 0;
  for (int kk = 0; kk < nk; ++kk) {
    const int nxt = cur ^ 1;
    if (kk + 1 < nk) {
      async_tile(Abase + (kk + 1) * 32, lda, smbase + nxt * 2 * TILEB, w, lane);
      async_tile(Bbase + (kk + 1) * 32, ldb, smbase + nxt * 2 * TILEB + TILEB, w, lane);
      // 8 outstanding; wait<=4 -> stage kk's 4 copies landed (in-order completion)
      asm volatile("s_wait_asynccnt 0x4" ::: "memory");
    } else {
      asm volatile("s_wait_asynccnt 0x0" ::: "memory");
    }
    __syncthreads();                        // stage kk visible to all waves

    const char* smA = smem + cur * 2 * TILEB;
    const char* smB = smem + cur * 2 * TILEB + TILEB;
    v16bf a0 = ldsA(smA, wm * 32,      lane);
    v16bf a1 = ldsA(smA, wm * 32 + 16, lane);
    v16bf b0 = ldsB(smB, wn * 64,      lane);
    v16bf b1 = ldsB(smB, wn * 64 + 16, lane);
    v16bf b2 = ldsB(smB, wn * 64 + 32, lane);
    v16bf b3 = ldsB(smB, wn * 64 + 48, lane);

    acc[0][0] = __builtin_amdgcn_wmma_f32_16x16x32_bf16(false, a0, false, b0, (short)0, acc[0][0], false, false);
    acc[0][1] = __builtin_amdgcn_wmma_f32_16x16x32_bf16(false, a0, false, b1, (short)0, acc[0][1], false, false);
    acc[0][2] = __builtin_amdgcn_wmma_f32_16x16x32_bf16(false, a0, false, b2, (short)0, acc[0][2], false, false);
    acc[0][3] = __builtin_amdgcn_wmma_f32_16x16x32_bf16(false, a0, false, b3, (short)0, acc[0][3], false, false);
    acc[1][0] = __builtin_amdgcn_wmma_f32_16x16x32_bf16(false, a1, false, b0, (short)0, acc[1][0], false, false);
    acc[1][1] = __builtin_amdgcn_wmma_f32_16x16x32_bf16(false, a1, false, b1, (short)0, acc[1][1], false, false);
    acc[1][2] = __builtin_amdgcn_wmma_f32_16x16x32_bf16(false, a1, false, b2, (short)0, acc[1][2], false, false);
    acc[1][3] = __builtin_amdgcn_wmma_f32_16x16x32_bf16(false, a1, false, b3, (short)0, acc[1][3], false, false);

    __syncthreads();                        // all waves done reading buf cur
    cur = nxt;
  }

  // C/D layout: lane&15 = N, VGPR r -> M = r + 8*(lane>>4)
  const int n = lane & 15, half = lane >> 4;
  const long cbase = (long)blockIdx.z * sC;
#pragma unroll
  for (int i = 0; i < 2; ++i)
#pragma unroll
    for (int j = 0; j < 4; ++j)
#pragma unroll
      for (int r = 0; r < 8; ++r) {
        float val = acc[i][j][r] * alpha;
        long idx = cbase + (row0 + wm * 32 + i * 16 + half * 8 + r) * (long)ldc
                         + (col0 + wn * 64 + j * 16 + n);
        if (Cf) Cf[idx] = val;
        else    Cb[idx] = (bf16_t)val;
      }
}

// ---------------- custom softmax: e/(1+sum e), in-place f32 row -> bf16 row --
__global__ void __launch_bounds__(256)
softmax_custom(float* __restrict__ att) {
  __shared__ float sm[256];
  float* x = att + (long)blockIdx.x * S_;
  bf16_t* y = (bf16_t*)x;                 // bf16 row overlays first half of f32 row
  int tid = threadIdx.x;
  float v[8];
  float lm = -3.4e38f;
#pragma unroll
  for (int i = 0; i < 8; ++i) { v[i] = x[i * 256 + tid]; lm = fmaxf(lm, v[i]); }
  sm[tid] = lm; __syncthreads();
  for (int s = 128; s > 0; s >>= 1) { if (tid < s) sm[tid] = fmaxf(sm[tid], sm[tid + s]); __syncthreads(); }
  float m = sm[0]; __syncthreads();
  float ls = 0.f;
#pragma unroll
  for (int i = 0; i < 8; ++i) { v[i] = __expf(v[i] - m); ls += v[i]; }
  sm[tid] = ls; __syncthreads();
  for (int s = 128; s > 0; s >>= 1) { if (tid < s) sm[tid] += sm[tid + s]; __syncthreads(); }
  float inv = 1.f / (1.f + sm[0]);        // the "+1" of the custom softmax
  __syncthreads();                        // all reads complete before in-place writes
#pragma unroll
  for (int i = 0; i < 8; ++i) y[i * 256 + tid] = (bf16_t)(v[i] * inv);
}

// ---------------- LayerNorm (D=512, 256 threads, 2 elems/thread) -> bf16 ----
__global__ void __launch_bounds__(256)
layernorm(const float* __restrict__ o, const float* __restrict__ g,
          const float* __restrict__ b, bf16_t* __restrict__ h) {
  __shared__ float sm[256];
  const float* x = o + (long)blockIdx.x * DIN;
  int tid = threadIdx.x;
  float x0 = x[tid], x1 = x[tid + 256];
  sm[tid] = x0 + x1; __syncthreads();
  for (int s = 128; s > 0; s >>= 1) { if (tid < s) sm[tid] += sm[tid + s]; __syncthreads(); }
  float mu = sm[0] * (1.f / DIN); __syncthreads();
  float d0 = x0 - mu, d1 = x1 - mu;
  sm[tid] = d0 * d0 + d1 * d1; __syncthreads();
  for (int s = 128; s > 0; s >>= 1) { if (tid < s) sm[tid] += sm[tid + s]; __syncthreads(); }
  float rstd = rsqrtf(sm[0] * (1.f / DIN) + LN_EPS);
  bf16_t* hr = h + (long)blockIdx.x * DIN;
  hr[tid]       = (bf16_t)(d0 * rstd * g[tid]       + b[tid]);
  hr[tid + 256] = (bf16_t)(d1 * rstd * g[tid + 256] + b[tid + 256]);
}

// ---------------------------------------------------------------------------
extern "C" void kernel_launch(void* const* d_in, const int* in_sizes, int n_in,
                              void* d_out, int out_size, void* d_ws, size_t ws_size,
                              hipStream_t stream) {
  (void)in_sizes; (void)n_in; (void)out_size;
  const float* x_st = (const float*)d_in[0];
  const float* x_fr = (const float*)d_in[1];
  const float* Wq   = (const float*)d_in[2];
  const float* Wk   = (const float*)d_in[3];
  const float* Wv   = (const float*)d_in[4];
  const float* Wo   = (const float*)d_in[5];
  const float* ln_g = (const float*)d_in[6];
  const float* ln_b = (const float*)d_in[7];
  const float* W1   = (const float*)d_in[8];
  const float* W2   = (const float*)d_in[9];
  float* out = (float*)d_out;

  const long MS = (long)B_ * S_;          // 32768 rows
  char* ws = (char*)d_ws;
  size_t off = 0;
  auto alloc = [&](size_t bytes) -> char* {
    char* p = ws + off; off += (bytes + 255) & ~(size_t)255; return p;
  };
  bf16_t* xst_b = (bf16_t*)alloc(MS * DIN * 2);
  bf16_t* xfr_b = (bf16_t*)alloc(MS * DIN * 2);
  bf16_t* WqT   = (bf16_t*)alloc((size_t)DIN * DIN * 2);
  bf16_t* WkT   = (bf16_t*)alloc((size_t)DIN * DIN * 2);
  bf16_t* WvT   = (bf16_t*)alloc((size_t)DIN * DIN * 2);
  bf16_t* WoT   = (bf16_t*)alloc((size_t)DIN * DIN * 2);
  bf16_t* W1T   = (bf16_t*)alloc((size_t)DIN * DFF * 2);
  bf16_t* W2T   = (bf16_t*)alloc((size_t)DFF * DIN * 2);
  bf16_t* qb    = (bf16_t*)alloc(MS * DIN * 2);
  bf16_t* kb    = (bf16_t*)alloc(MS * DIN * 2);
  bf16_t* vb    = (bf16_t*)alloc(MS * DIN * 2);
  bf16_t* vtb   = (bf16_t*)alloc(MS * DIN * 2);
  float*  att   = (float*) alloc((size_t)B_ * S_ * S_ * 4);   // score bf16 in-place
  bf16_t* oat   = (bf16_t*)alloc(MS * DIN * 2);
  float*  opr   = (float*) alloc(MS * DIN * 4);
  bf16_t* hb    = (bf16_t*)alloc(MS * DIN * 2);
  bf16_t* t1    = (bf16_t*)alloc(MS * DFF * 2);
  if (off > ws_size) return;              // insufficient workspace

  const dim3 blk(256);
  const int nx = (int)(MS * DIN);

  // precision conversion + weight transposes (one-time cost, tiny vs GEMMs)
  cvt_f32_bf16<<<dim3((nx + 255) / 256), blk, 0, stream>>>(x_st, xst_b, nx);
  cvt_f32_bf16<<<dim3((nx + 255) / 256), blk, 0, stream>>>(x_fr, xfr_b, nx);
  wtrans<<<dim3((DIN * DIN + 255) / 256), blk, 0, stream>>>(Wq, WqT, DIN, DIN);
  wtrans<<<dim3((DIN * DIN + 255) / 256), blk, 0, stream>>>(Wk, WkT, DIN, DIN);
  wtrans<<<dim3((DIN * DIN + 255) / 256), blk, 0, stream>>>(Wv, WvT, DIN, DIN);
  wtrans<<<dim3((DIN * DIN + 255) / 256), blk, 0, stream>>>(Wo, WoT, DIN, DIN);
  wtrans<<<dim3((DIN * DFF + 255) / 256), blk, 0, stream>>>(W1, W1T, DIN, DFF);
  wtrans<<<dim3((DFF * DIN + 255) / 256), blk, 0, stream>>>(W2, W2T, DFF, DIN);

  // q/k/v projections: [32768x512] = [32768x512] * [512x512]
  dim3 g_qkv(DIN / 128, MS / 128, 1);
  gemm_bf16<<<g_qkv, blk, 0, stream>>>(xst_b, WqT, nullptr, qb, DIN, DIN, DIN, DIN, 0, 0, 0, 1.f);
  gemm_bf16<<<g_qkv, blk, 0, stream>>>(xfr_b, WkT, nullptr, kb, DIN, DIN, DIN, DIN, 0, 0, 0, 1.f);
  gemm_bf16<<<g_qkv, blk, 0, stream>>>(xfr_b, WvT, nullptr, vb, DIN, DIN, DIN, DIN, 0, 0, 0, 1.f);
  vtrans<<<dim3((unsigned)(MS * DIN / 256)), blk, 0, stream>>>(vb, vtb);

  // att = q*k^T / 256  (batched over B; k row-major already IS the [N][K] operand)
  dim3 g_att(S_ / 128, S_ / 128, B_);
  gemm_bf16<<<g_att, blk, 0, stream>>>(qb, kb, att, nullptr, DIN, DIN, DIN, S_,
                                       (long)S_ * DIN, (long)S_ * DIN, (long)S_ * S_,
                                       1.f / 256.f);
  // custom softmax, f32 row -> bf16 row in place
  softmax_custom<<<dim3((unsigned)MS), blk, 0, stream>>>(att);

  // o = score * v   (A = bf16 score rows with stride 2*S over the f32 buffer)
  dim3 g_o(DIN / 128, S_ / 128, B_);
  gemm_bf16<<<g_o, blk, 0, stream>>>((const bf16_t*)att, vtb, nullptr, oat,
                                     S_, 2 * S_, S_, DIN,
                                     (long)2 * S_ * S_, (long)DIN * S_, (long)S_ * DIN,
                                     1.f);
  // o @ Wo -> f32
  dim3 g_op(DIN / 128, MS / 128, 1);
  gemm_bf16<<<g_op, blk, 0, stream>>>(oat, WoT, opr, nullptr, DIN, DIN, DIN, DIN, 0, 0, 0, 1.f);

  // LayerNorm -> bf16
  layernorm<<<dim3((unsigned)MS), blk, 0, stream>>>(opr, ln_g, ln_b, hb);

  // t1 = h @ W1 : [32768x2048]
  dim3 g_t1(DFF / 128, MS / 128, 1);
  gemm_bf16<<<g_t1, blk, 0, stream>>>(hb, W1T, nullptr, t1, DIN, DIN, DIN, DFF, 0, 0, 0, 1.f);

  // out = t1 @ W2 : [32768x512] f32 -> d_out
  dim3 g_out(DIN / 128, MS / 128, 1);
  gemm_bf16<<<g_out, blk, 0, stream>>>(t1, W2T, out, nullptr, DFF, DFF, DFF, DIN, 0, 0, 0, 1.f);
}
